// Block_46703474377400
// MI455X (gfx1250) — matrix operand order repaired and checked by reading, not compiled, not run on hardware
//
#include <hip/hip_runtime.h>

// ---------------------------------------------------------------------------
// Transformer block for MI455X (gfx1250, wave32, WMMA bf16 -> f32 accum)
// B=2, S=2048, E=1024, H=16, DH=64
// Round 2: 128x128 GEMM tiles (8 WMMA/wave/k-step), async global->LDS staging
// (global_load_async_to_lds_b128 + s_wait_asynccnt), global prefetch.
// ---------------------------------------------------------------------------

#define B_  2
#define S_  2048
#define E_  1024
#define H_  16
#define DH_ 64
#define M_  (B_ * S_)      // 4096 rows
#define FF_ (4 * E_)       // 4096

typedef __attribute__((ext_vector_type(16))) __bf16 v16bf;
typedef __attribute__((ext_vector_type(8)))  float  v8f;

#define WMMA_BF16(a, b, c) \
  __builtin_amdgcn_wmma_f32_16x16x32_bf16(false, (a), false, (b), (short)0, (c), false, false)

__device__ __forceinline__ unsigned short f2bf(float f) {
  unsigned u = __float_as_uint(f);
  u += 0x7FFFu + ((u >> 16) & 1u);     // round-to-nearest-even
  return (unsigned short)(u >> 16);
}

// Async 16-byte copy global -> LDS (GVS mode: saddr base + 32-bit vgpr byte
// offset). Tracked by ASYNCcnt; data bypasses VGPRs entirely.
__device__ __forceinline__ void async_copy_b128(const void* gbase, unsigned gbyteoff,
                                                void* lds_ptr) {
  unsigned lds_addr = (unsigned)(uintptr_t)lds_ptr;   // low 32 bits = LDS offset
  asm volatile("global_load_async_to_lds_b128 %0, %1, %2"
               :: "v"(lds_addr), "v"(gbyteoff), "s"(gbase)
               : "memory");
}
__device__ __forceinline__ void wait_async0() {
  asm volatile("s_wait_asynccnt 0x0" ::: "memory");
}

// A fragment: 16x32 bf16, row-major source, row stride ld (elements).
// lanes 0-15: row=lane, K=0..15 (2 per VGPR); lanes 16-31: same rows, K=16..31
// (contiguous order, matching load_frag_b)
__device__ __forceinline__ v16bf load_frag_a(const unsigned short* base, int ld) {
  const int lane = threadIdx.x & 31;
  const unsigned short* p = base + (size_t)(lane & 15) * ld + 16 * (lane >> 4);
  union { v16bf v; unsigned u[8]; } r;
#pragma unroll
  for (int i = 0; i < 8; ++i)
    r.u[i] = *reinterpret_cast<const unsigned*>(p + 2 * i);
  return r.v;
}

// B fragment: 32x16 bf16 (KxN). Source stored N-major: Bt[n][k], row stride ld.
// lanes 0-15 hold K=0..15 (2 per VGPR), lanes 16-31 hold K=16..31; lane -> N col.
__device__ __forceinline__ v16bf load_frag_b(const unsigned short* base, int ld) {
  const int lane = threadIdx.x & 31;
  const unsigned short* p = base + (size_t)(lane & 15) * ld + 16 * (lane >> 4);
  union { v16bf v; unsigned u[8]; } r;
#pragma unroll
  for (int i = 0; i < 8; ++i)
    r.u[i] = *reinterpret_cast<const unsigned*>(p + 2 * i);
  return r.v;
}

// ---------------------------------------------------------------------------
// f32 -> bf16 conversion (weights)
// ---------------------------------------------------------------------------
__global__ __launch_bounds__(256) void cvt_bf16_kernel(const float* __restrict__ in,
                                                       unsigned short* __restrict__ out,
                                                       int n) {
  int i = blockIdx.x * 256 + threadIdx.x;
  if (i < n) out[i] = f2bf(in[i]);
}

// ---------------------------------------------------------------------------
// Fused LayerNorm -> bf16. One block per row of E_=1024 (4 elems / thread).
// ---------------------------------------------------------------------------
__global__ __launch_bounds__(256) void layernorm_bf16_kernel(const float* __restrict__ x,
                                                             const float* __restrict__ gamma,
                                                             const float* __restrict__ beta,
                                                             unsigned short* __restrict__ out) {
  const int row = blockIdx.x;
  const int tid = threadIdx.x;
  const float* xr = x + (size_t)row * E_;
  float v[4], s = 0.f, ss = 0.f;
#pragma unroll
  for (int j = 0; j < 4; ++j) {
    float t = xr[tid + j * 256];
    v[j] = t; s += t; ss += t * t;
  }
#pragma unroll
  for (int m = 1; m < 32; m <<= 1) {
    s  += __shfl_xor(s, m, 32);
    ss += __shfl_xor(ss, m, 32);
  }
  __shared__ float sb[8], ssb[8];
  if ((tid & 31) == 0) { sb[tid >> 5] = s; ssb[tid >> 5] = ss; }
  __syncthreads();
  s = 0.f; ss = 0.f;
#pragma unroll
  for (int i = 0; i < 8; ++i) { s += sb[i]; ss += ssb[i]; }
  const float mean = s * (1.0f / E_);
  const float var  = ss * (1.0f / E_) - mean * mean;
  const float rstd = rsqrtf(var + 1e-5f);
#pragma unroll
  for (int j = 0; j < 4; ++j) {
    int c = tid + j * 256;
    out[(size_t)row * E_ + c] = f2bf((v[j] - mean) * rstd * gamma[c] + beta[c]);
  }
}

// ---------------------------------------------------------------------------
// Generic bf16 GEMM: C[M,N] = A[M,K] * B[K,N] (+bias)(+ReLU)(+residual)
// Block tile 128x128, 8 waves in a 4x2 grid, each wave 32x64 (2x4 fragments,
// 8 WMMAs per K-step of 32). A staged via async-to-LDS B128 copies; B staged
// N-major (transposed) via VALU so B-fragment loads are contiguous b128s.
// ---------------------------------------------------------------------------
template <bool RELU, bool RESID, bool OUTBF16>
__global__ __launch_bounds__(256) void gemm_bf16_kernel(const unsigned short* __restrict__ A,
                                                        const unsigned short* __restrict__ Bw,
                                                        const float* __restrict__ bias,
                                                        const float* __restrict__ resid,
                                                        void* __restrict__ out,
                                                        int M, int N, int K) {
  __shared__ unsigned short As[128][40];   // 128 x 32 (+pad, 80B row: 16B aligned)
  __shared__ unsigned short Bt[128][40];   // N-major: Bt[n][k]
  const int tid  = threadIdx.x;
  const int wave = tid >> 5;
  const int lane = tid & 31;
  const int wm   = wave >> 1;              // 0..3  -> 32-row slice
  const int wn   = wave & 1;               // 0..1  -> 64-col slice
  const int m0 = blockIdx.x * 128;
  const int n0 = blockIdx.y * 128;

  v8f acc[2][4];
#pragma unroll
  for (int s = 0; s < 2; ++s)
#pragma unroll
    for (int g = 0; g < 4; ++g)
#pragma unroll
      for (int r = 0; r < 8; ++r) acc[s][g][r] = 0.f;

  for (int k0 = 0; k0 < K; k0 += 32) {
    __syncthreads();
    // A tile: 128x32 bf16 = 512 x 16B chunks; 2 async copies per thread.
#pragma unroll
    for (int j = 0; j < 2; ++j) {
      int q   = tid + j * 256;           // 0..511
      int r   = q >> 2;                  // row 0..127
      int seg = q & 3;                   // 16B segment within 64B row
      async_copy_b128(A, (unsigned)((((size_t)(m0 + r) * K) + k0 + seg * 8) * 2),
                      &As[r][seg * 8]);
    }
    // B tile: 32x128, read pairs along N, scatter transposed into Bt[n][k]
#pragma unroll
    for (int j = 0; j < 8; ++j) {
      int q  = tid + j * 256;            // 0..2047
      int kk = q >> 6;                   // 0..31
      int np = (q & 63) << 1;            // 0..126 even
      unsigned d = *reinterpret_cast<const unsigned*>(Bw + (size_t)(k0 + kk) * N + n0 + np);
      Bt[np][kk]     = (unsigned short)(d & 0xFFFFu);
      Bt[np + 1][kk] = (unsigned short)(d >> 16);
    }
    // prefetch next B tile into cache while we compute
    if (k0 + 32 < K)
      __builtin_prefetch(Bw + (size_t)(k0 + 32 + (tid >> 3)) * N + n0 + ((tid & 7) << 4), 0, 1);
    wait_async0();                       // this wave's async LDS writes done
    __syncthreads();                     // publish block-wide

    v16bf a0 = load_frag_a(&As[wm * 32][0], 40);
    v16bf a1 = load_frag_a(&As[wm * 32 + 16][0], 40);
#pragma unroll
    for (int g = 0; g < 4; ++g) {
      v16bf b = load_frag_b(&Bt[wn * 64 + g * 16][0], 40);
      acc[0][g] = WMMA_BF16(a0, b, acc[0][g]);
      acc[1][g] = WMMA_BF16(a1, b, acc[1][g]);
    }
  }

  const int col = lane & 15, hl = lane >> 4;
#pragma unroll
  for (int s = 0; s < 2; ++s) {
#pragma unroll
    for (int g = 0; g < 4; ++g) {
      int n = n0 + wn * 64 + g * 16 + col;
      float bv = bias ? bias[n] : 0.f;
#pragma unroll
      for (int r = 0; r < 8; ++r) {
        int m = m0 + wm * 32 + s * 16 + r + 8 * hl;
        float v = acc[s][g][r] + bv;
        if (RELU)  v = v > 0.f ? v : 0.f;
        if (RESID) v += resid[(size_t)m * N + n];
        if (OUTBF16)
          reinterpret_cast<unsigned short*>(out)[(size_t)m * N + n] = f2bf(v);
        else
          reinterpret_cast<float*>(out)[(size_t)m * N + n] = v;
      }
    }
  }
}

// ---------------------------------------------------------------------------
// Per-head QKV projection: out[b,h,s,:] = h1[b*S+s,:] @ W[h]  (N = DH = 64)
// Block tile 128x64, wave -> 16 rows x 64 cols.
// ---------------------------------------------------------------------------
__global__ __launch_bounds__(256) void gemm_head_kernel(const unsigned short* __restrict__ A,
                                                        const unsigned short* __restrict__ W,
                                                        unsigned short* __restrict__ out,
                                                        int M, int K) {
  __shared__ unsigned short As[128][40];
  __shared__ unsigned short Bt[64][40];
  const int tid  = threadIdx.x;
  const int wave = tid >> 5;
  const int lane = tid & 31;
  const int m0   = blockIdx.x * 128;
  const int head = blockIdx.z;
  const unsigned short* Bw = W + (size_t)head * K * DH_;

  v8f acc[4];
#pragma unroll
  for (int g = 0; g < 4; ++g)
#pragma unroll
    for (int r = 0; r < 8; ++r) acc[g][r] = 0.f;

  for (int k0 = 0; k0 < K; k0 += 32) {
    __syncthreads();
#pragma unroll
    for (int j = 0; j < 2; ++j) {       // async A tile: 512 x 16B chunks
      int q   = tid + j * 256;
      int r   = q >> 2;
      int seg = q & 3;
      async_copy_b128(A, (unsigned)((((size_t)(m0 + r) * K) + k0 + seg * 8) * 2),
                      &As[r][seg * 8]);
    }
#pragma unroll
    for (int j = 0; j < 4; ++j) {       // B tile 32x64 transposed
      int q  = tid + j * 256;
      int kk = q >> 5;
      int np = (q & 31) << 1;
      unsigned d = *reinterpret_cast<const unsigned*>(Bw + (size_t)(k0 + kk) * DH_ + np);
      Bt[np][kk]     = (unsigned short)(d & 0xFFFFu);
      Bt[np + 1][kk] = (unsigned short)(d >> 16);
    }
    wait_async0();
    __syncthreads();

    v16bf a = load_frag_a(&As[wave * 16][0], 40);
#pragma unroll
    for (int g = 0; g < 4; ++g) {
      v16bf b = load_frag_b(&Bt[g * 16][0], 40);
      acc[g] = WMMA_BF16(a, b, acc[g]);
    }
  }

  const int col = lane & 15, hl = lane >> 4;
#pragma unroll
  for (int g = 0; g < 4; ++g) {
    int n = g * 16 + col;
#pragma unroll
    for (int r = 0; r < 8; ++r) {
      int m = m0 + wave * 16 + r + 8 * hl;
      int b = m / S_, s = m % S_;
      out[(((size_t)b * H_ + head) * S_ + s) * DH_ + n] = f2bf(acc[g][r]);
    }
  }
}

// ---------------------------------------------------------------------------
// Flash attention (no causal mask, matching the reference).
// Block = 128 query rows (8 waves x 16). Key tiles of 32; K tile async-staged
// row-major in LDS (QK^T B-frags contiguous), V tile transposed in LDS (PV
// B-frags contiguous). Online softmax; P round-trips through per-wave LDS.
// ---------------------------------------------------------------------------
__global__ __launch_bounds__(256) void flash_attn_kernel(const unsigned short* __restrict__ Q,
                                                         const unsigned short* __restrict__ Kd,
                                                         const unsigned short* __restrict__ Vd,
                                                         unsigned short* __restrict__ attn) {
  __shared__ unsigned short Kt[32][72];       // [key][dh] (144B row: 16B aligned)
  __shared__ unsigned short Vt[64][40];       // [dh][key] (transposed)
  __shared__ unsigned short Pt[8][16][40];    // per-wave P tile 16x32

  const int tid  = threadIdx.x;
  const int wave = tid >> 5;
  const int lane = tid & 31;
  const int col  = lane & 15, hl = lane >> 4;
  const int b = blockIdx.z, h = blockIdx.y;
  const int q0 = blockIdx.x * 128 + wave * 16;
  const size_t bh = ((size_t)b * H_ + h) * (size_t)S_ * DH_;

  const v16bf qf0 = load_frag_a(Q + bh + (size_t)q0 * DH_, DH_);
  const v16bf qf1 = load_frag_a(Q + bh + (size_t)q0 * DH_ + 32, DH_);

  v8f o[4];
  float mr[8], lr[8];
#pragma unroll
  for (int g = 0; g < 4; ++g)
#pragma unroll
    for (int r = 0; r < 8; ++r) o[g][r] = 0.f;
#pragma unroll
  for (int r = 0; r < 8; ++r) { mr[r] = -1e30f; lr[r] = 0.f; }

  for (int kt = 0; kt < S_; kt += 32) {
    __syncthreads();
    {                                    // K tile 32x64: 256 x 16B async chunks
      int r   = tid >> 3;
      int seg = tid & 7;
      async_copy_b128(Kd, (unsigned)((bh + (size_t)(kt + r) * DH_ + seg * 8) * 2),
                      &Kt[r][seg * 8]);
    }
#pragma unroll
    for (int j = 0; j < 4; ++j) {        // V tile: 32x64 -> transposed
      int q  = tid + j * 256;
      int r  = q >> 5;
      int dp = (q & 31) << 1;
      unsigned d = *reinterpret_cast<const unsigned*>(Vd + bh + (size_t)(kt + r) * DH_ + dp);
      Vt[dp][r]     = (unsigned short)(d & 0xFFFFu);
      Vt[dp + 1][r] = (unsigned short)(d >> 16);
    }
    wait_async0();
    __syncthreads();

    // scores: S = Q (16x64) x K^T -> two 16x16 fragments (keys 0-15, 16-31)
    v8f s0, s1;
#pragma unroll
    for (int r = 0; r < 8; ++r) { s0[r] = 0.f; s1[r] = 0.f; }
    {
      v16bf b00 = load_frag_b(&Kt[0][0],  72);
      v16bf b01 = load_frag_b(&Kt[0][32], 72);
      v16bf b10 = load_frag_b(&Kt[16][0], 72);
      v16bf b11 = load_frag_b(&Kt[16][32],72);
      s0 = WMMA_BF16(qf0, b00, s0);
      s0 = WMMA_BF16(qf1, b01, s0);
      s1 = WMMA_BF16(qf0, b10, s1);
      s1 = WMMA_BF16(qf1, b11, s1);
    }

    // online softmax (row reductions across the 16 lanes of each half)
#pragma unroll
    for (int r = 0; r < 8; ++r) {
      float a0 = s0[r] * 0.125f;        // DH^-0.5 = 1/8
      float a1 = s1[r] * 0.125f;
      float t = fmaxf(a0, a1);
      t = fmaxf(t, __shfl_xor(t, 1, 32));
      t = fmaxf(t, __shfl_xor(t, 2, 32));
      t = fmaxf(t, __shfl_xor(t, 4, 32));
      t = fmaxf(t, __shfl_xor(t, 8, 32));
      float mn = fmaxf(mr[r], t);
      float al = __expf(mr[r] - mn);
      mr[r] = mn;
      float p0 = __expf(a0 - mn);
      float p1 = __expf(a1 - mn);
      int pr = r + 8 * hl;
      Pt[wave][pr][col]      = f2bf(p0);
      Pt[wave][pr][col + 16] = f2bf(p1);
      float rs = p0 + p1;
      rs += __shfl_xor(rs, 1, 32);
      rs += __shfl_xor(rs, 2, 32);
      rs += __shfl_xor(rs, 4, 32);
      rs += __shfl_xor(rs, 8, 32);
      lr[r] = lr[r] * al + rs;
#pragma unroll
      for (int g = 0; g < 4; ++g) o[g][r] *= al;
    }

    // O += P (16x32) x V (32x64)
    v16bf pf = load_frag_a(&Pt[wave][0][0], 40);
#pragma unroll
    for (int g = 0; g < 4; ++g) {
      v16bf vb = load_frag_b(&Vt[g * 16][0], 40);
      o[g] = WMMA_BF16(pf, vb, o[g]);
    }
  }

  // finalize: O /= l, write attn[b, s, h*64 + d] as bf16
#pragma unroll
  for (int r = 0; r < 8; ++r) {
    float inv = lr[r] > 0.f ? 1.0f / lr[r] : 0.f;
    int m = q0 + r + 8 * hl;
#pragma unroll
    for (int g = 0; g < 4; ++g) {
      attn[((size_t)b * S_ + m) * (size_t)(H_ * DH_) + h * DH_ + g * 16 + col] =
          f2bf(o[g][r] * inv);
    }
  }
}

// ---------------------------------------------------------------------------
// Launch
// ---------------------------------------------------------------------------
extern "C" void kernel_launch(void* const* d_in, const int* in_sizes, int n_in,
                              void* d_out, int out_size, void* d_ws, size_t ws_size,
                              hipStream_t stream) {
  (void)in_sizes; (void)n_in; (void)out_size; (void)ws_size;
  const float* x     = (const float*)d_in[0];
  const float* Wq    = (const float*)d_in[1];
  const float* Wk    = (const float*)d_in[2];
  const float* Wv    = (const float*)d_in[3];
  const float* Wproj = (const float*)d_in[4];
  const float* bproj = (const float*)d_in[5];
  const float* W1    = (const float*)d_in[6];
  const float* b1    = (const float*)d_in[7];
  const float* W2    = (const float*)d_in[8];
  const float* b2    = (const float*)d_in[9];
  const float* g1    = (const float*)d_in[10];
  const float* be1   = (const float*)d_in[11];
  const float* g2    = (const float*)d_in[12];
  const float* be2   = (const float*)d_in[13];

  char* ws = (char*)d_ws;
  size_t off = 0;
  auto alloc = [&](size_t bytes) -> void* {
    void* p = ws + off;
    off += (bytes + 255) & ~(size_t)255;
    return p;
  };
  unsigned short* h1   = (unsigned short*)alloc((size_t)M_ * E_ * 2);
  unsigned short* Wqb  = (unsigned short*)alloc((size_t)H_ * E_ * DH_ * 2);
  unsigned short* Wkb  = (unsigned short*)alloc((size_t)H_ * E_ * DH_ * 2);
  unsigned short* Wvb  = (unsigned short*)alloc((size_t)H_ * E_ * DH_ * 2);
  unsigned short* Wpb  = (unsigned short*)alloc((size_t)E_ * E_ * 2);
  unsigned short* W1b  = (unsigned short*)alloc((size_t)E_ * FF_ * 2);
  unsigned short* W2b  = (unsigned short*)alloc((size_t)FF_ * E_ * 2);
  unsigned short* Qb   = (unsigned short*)alloc((size_t)M_ * E_ * 2);
  unsigned short* Kb   = (unsigned short*)alloc((size_t)M_ * E_ * 2);
  unsigned short* Vb   = (unsigned short*)alloc((size_t)M_ * E_ * 2);
  unsigned short* attn = (unsigned short*)alloc((size_t)M_ * E_ * 2);
  float*          x1   = (float*)alloc((size_t)M_ * E_ * 4);
  unsigned short* h2   = (unsigned short*)alloc((size_t)M_ * E_ * 2);
  unsigned short* ffn1 = (unsigned short*)alloc((size_t)M_ * FF_ * 2);

  const dim3 blk(256);
  auto cvt = [&](const float* src, unsigned short* dst, int n) {
    cvt_bf16_kernel<<<dim3((n + 255) / 256), blk, 0, stream>>>(src, dst, n);
  };
  cvt(Wq, Wqb, H_ * E_ * DH_);
  cvt(Wk, Wkb, H_ * E_ * DH_);
  cvt(Wv, Wvb, H_ * E_ * DH_);
  cvt(Wproj, Wpb, E_ * E_);
  cvt(W1, W1b, E_ * FF_);
  cvt(W2, W2b, FF_ * E_);

  // LN1
  layernorm_bf16_kernel<<<dim3(M_), blk, 0, stream>>>(x, g1, be1, h1);

  // QKV per-head projections
  gemm_head_kernel<<<dim3(M_ / 128, 1, H_), blk, 0, stream>>>(h1, Wqb, Qb, M_, E_);
  gemm_head_kernel<<<dim3(M_ / 128, 1, H_), blk, 0, stream>>>(h1, Wkb, Kb, M_, E_);
  gemm_head_kernel<<<dim3(M_ / 128, 1, H_), blk, 0, stream>>>(h1, Wvb, Vb, M_, E_);

  // attention
  flash_attn_kernel<<<dim3(S_ / 128, H_, B_), blk, 0, stream>>>(Qb, Kb, Vb, attn);

  // x1 = x + attn @ Wproj + bproj
  gemm_bf16_kernel<false, true, false><<<dim3(M_ / 128, E_ / 128), blk, 0, stream>>>(
      attn, Wpb, bproj, x, (void*)x1, M_, E_, E_);

  // LN2
  layernorm_bf16_kernel<<<dim3(M_), blk, 0, stream>>>(x1, g2, be2, h2);

  // ffn1 = relu(h2 @ W1 + b1)
  gemm_bf16_kernel<true, false, true><<<dim3(M_ / 128, FF_ / 128), blk, 0, stream>>>(
      h2, W1b, b1, nullptr, (void*)ffn1, M_, FF_, E_);

  // out = x1 + ffn1 @ W2 + b2
  gemm_bf16_kernel<false, true, false><<<dim3(M_ / 128, E_ / 128), blk, 0, stream>>>(
      ffn1, W2b, b2, x1, d_out, M_, E_, FF_);
}
